// Decoder_14894946582933
// MI455X (gfx1250) — compile-verified, matrix operand order checked by reading, not run on hardware
//
#include <hip/hip_runtime.h>

#define TSN 16
#define BN  4
#define CN  256
#define HN  48
#define WN  48
#define NIMG (TSN*BN)                 // 64 images after (ts,b) fold
#define KTOT (CN*9)                   // 2304 = reduction dim of 3x3 conv
#define WS_WH 1179648UL               // byte offset of head-weight pack in ws
#define WS_Y  1196032UL               // byte offset of Y (conv output) in ws
#define BOXELT ((size_t)TSN*BN*12*HN*WN)  // #elements of boxes output

typedef __attribute__((ext_vector_type(16))) __bf16 v16bf;
typedef __attribute__((ext_vector_type(8)))  __bf16 v8bf;
typedef __attribute__((ext_vector_type(8)))  float  v8f;
typedef __attribute__((ext_vector_type(4)))  unsigned int v4u;
typedef __attribute__((ext_vector_type(8)))  int v8i;
typedef __attribute__((ext_vector_type(4)))  int v4i;

union ABu { v16bf v; v8bf h[2]; };

__device__ __forceinline__ unsigned short f2bf(float f) {
  union { float f; unsigned int u; } x; x.f = f;
  unsigned int u = x.u;
  return (unsigned short)((u + 0x7FFFu + ((u >> 16) & 1u)) >> 16);  // RNE
}

// Issue one TDM load: 2D bf16 tile [256 rows x 256 elems], row stride 2304 elems,
// from global `gsrc` into LDS byte offset `lds_off`. D# per CDNA5 ISA ch.8.
__device__ __forceinline__ void tdm_load_w_tap(const unsigned short* gsrc, unsigned lds_off) {
  unsigned long long ga = (unsigned long long)(size_t)gsrc;
  v4u g0 = { 1u,                                     // count=1, user mode
             lds_off,                                // lds_addr (bits 63:32)
             (unsigned)(ga & 0xFFFFFFFFu),           // global_addr[31:0]
             ((unsigned)((ga >> 32) & 0x01FFFFFFu))  // global_addr[56:32]
               | 0x80000000u };                      // type=2 ("image")
  v8i g1 = { 0x00010000,   // data_size=1 (2 bytes), no multicast/pad/iterate
             0x09000000,   // tensor_dim0 = 2304 (low16 << 16)
             0x01000000,   // tensor_dim1 = 256 (low16 << 16)
             0x01000000,   // tile_dim0   = 256 (<< 16)
             0x00000100,   // tile_dim1   = 256, tile_dim2 = 0
             2304,         // tensor_dim0_stride[31:0]
             0, 0 };       // stride hi / tensor_dim1_stride
  v4i z4 = { 0, 0, 0, 0 };
#if __has_include(<hip/amd_detail/amd_gfx1250_TDM.h>)
  v8i z8 = { 0, 0, 0, 0, 0, 0, 0, 0 };
  __builtin_amdgcn_tensor_load_to_lds(g0, g1, z4, z4, z8, 0);
#else
  __builtin_amdgcn_tensor_load_to_lds(g0, g1, z4, z4, 0);
#endif
}

// ---- prep: conv weights (cout,cin,ky,kx) f32 -> bf16 [cout][tap*256+cin] ----
__global__ __launch_bounds__(256)
void prep_conv_w(const float* __restrict__ w, unsigned short* __restrict__ Bw) {
  int idx = blockIdx.x * 256 + threadIdx.x;
  if (idx >= CN * CN * 9) return;
  int kx   = idx % 3;
  int ky   = (idx / 3) % 3;
  int cin  = (idx / 9) & 255;
  int cout = idx / (9 * CN);
  Bw[(size_t)cout * KTOT + (ky * 3 + kx) * CN + cin] = f2bf(w[idx]);
}

// ---- prep: head weights -> bf16 [32][256] (rows 0..11 box, 12..20 cls, rest 0)
__global__ __launch_bounds__(256)
void prep_head_w(const float* __restrict__ bw, const float* __restrict__ cw,
                 unsigned short* __restrict__ Wh) {
  int idx = blockIdx.x * 256 + threadIdx.x;   // < 8192
  int o = idx >> 8, c = idx & 255;
  float v = 0.f;
  if (o < 12)      v = bw[o * CN + c];
  else if (o < 21) v = cw[(o - 12) * CN + c];
  Wh[idx] = f2bf(v);
}

// ---- 3x3 SAME conv as implicit GEMM, bf16 WMMA, f32 accumulate ------------
// block = 256 threads (8 waves); one block computes 2 output rows x 256 couts.
// Weights staged per-tap into LDS by the Tensor Data Mover.
__global__ __launch_bounds__(256)
void conv3x3_wmma(const float* __restrict__ X, const unsigned short* __restrict__ Bw,
                  const float* __restrict__ conv_b, float* __restrict__ Y) {
  __shared__ __align__(16) unsigned short Xs[4 * 50 * CN];  // 100 KB input stage (bf16)
  __shared__ __align__(16) unsigned short Wt[CN * CN];      // 128 KB per-tap weight slab

  const int tid = threadIdx.x;
  const int n   = blockIdx.x / 24;
  const int y0  = (blockIdx.x % 24) * 2;

  // stage input (f32 -> bf16), coalesced over x
  for (int j = 0; j < 192; ++j) {
    int f   = j * 256 + tid;            // over (r, cin, xx)
    int xx  = f % 48;
    int cin = (f / 48) & 255;
    int r   = f / (48 * 256);           // 0..3
    int yy  = y0 - 1 + r;
    float val = 0.f;
    if (yy >= 0 && yy < HN)
      val = X[(((size_t)n * CN + cin) * HN + yy) * WN + xx];
    Xs[(r * 50 + xx + 1) * CN + cin] = f2bf(val);
  }
  for (int j = 0; j < 8; ++j) {         // zero halo columns 0 and 49
    int f = j * 256 + tid;
    int cin = f & 255;
    int rc  = f >> 8;                   // 0..7
    int col = (rc & 1) ? 49 : 0;
    Xs[((rc >> 1) * 50 + col) * CN + cin] = 0;
  }

  const int lane = tid & 31, wid = tid >> 5;
  const int l16 = lane & 15, hi = lane >> 4;
  const int wM  = wid >> 1;             // cout group of 64
  const int wN  = wid & 1;              // which of the 2 output rows
  const int klo = hi * 8;               // K-offset per A/B 16-bit VGPR striping
  const unsigned wt_off = (unsigned)(size_t)(void*)&Wt[0];  // LDS byte offset

  v8f acc[4][3];
  #pragma unroll
  for (int m = 0; m < 4; ++m)
    #pragma unroll
    for (int p = 0; p < 3; ++p) {
      v8f z = {0.f,0.f,0.f,0.f,0.f,0.f,0.f,0.f};
      acc[m][p] = z;
    }

  for (int tap = 0; tap < 9; ++tap) {
    const int dy = tap / 3 - 1, dx = tap % 3 - 1;
    const int r  = wN + dy + 1;         // 0..3 LDS row

    __syncthreads();                    // prior tap's Wt reads done (+Xs staged)
    if (tid == 0) {                     // TDM: weights for this tap -> LDS
      tdm_load_w_tap(Bw + (size_t)tap * CN, wt_off);
      __builtin_amdgcn_s_wait_tensorcnt(0);
    }
    __syncthreads();                    // Wt visible to all waves

    #pragma unroll
    for (int ch = 0; ch < 8; ++ch) {
      const int cin0 = ch * 32;
      // B fragments: X tile [32k x 16px], from LDS (two 16B ds loads per lane)
      ABu bf[3];
      #pragma unroll
      for (int p = 0; p < 3; ++p) {
        const v8bf* q = (const v8bf*)&Xs[(r * 50 + p * 16 + l16 + dx + 1) * CN + cin0 + klo];
        bf[p].h[0] = q[0];
        bf[p].h[1] = q[2];              // +16 elements (K+16)
      }
      // A fragments: weights [16cout x 32k], from the TDM-staged LDS slab
      #pragma unroll
      for (int m = 0; m < 4; ++m) {
        const int cout = wM * 64 + m * 16 + l16;
        const v8bf* pa = (const v8bf*)&Wt[cout * CN + cin0 + klo];
        ABu af; af.h[0] = pa[0]; af.h[1] = pa[2];
        #pragma unroll
        for (int p = 0; p < 3; ++p)
          acc[m][p] = __builtin_amdgcn_wmma_f32_16x16x32_bf16(
              false, af.v, false, bf[p].v, (short)0, acc[m][p], false, false);
      }
    }
  }

  // epilogue: +bias, store Y[n][cout][y][x]; per-VGPR rows are 2x64B contiguous
  const int yrow = y0 + wN;
  #pragma unroll
  for (int m = 0; m < 4; ++m) {
    const int coutB = wM * 64 + m * 16 + hi * 8;
    #pragma unroll
    for (int p = 0; p < 3; ++p) {
      const int px = p * 16 + l16;
      const size_t base = (((size_t)n * CN + coutB) * HN + yrow) * WN + px;
      #pragma unroll
      for (int vv = 0; vv < 8; ++vv)
        Y[base + (size_t)vv * HN * WN] = acc[m][p][vv] + conv_b[coutB + vv];
    }
  }
}

// ---- fused LI scan (registers) + 1x1 heads (WMMA bf16) --------------------
// block handles one (b, y) row: 256c x 48px state in registers across t
__global__ __launch_bounds__(256)
void li_scan_heads(const float* __restrict__ Y, const unsigned short* __restrict__ Wh,
                   const float* __restrict__ box_b, const float* __restrict__ cls_b,
                   float* __restrict__ out) {
  __shared__ __align__(16) unsigned short Vs[WN * CN];   // [px][c] bf16
  __shared__ __align__(16) unsigned short Whs[32 * CN];  // [o][c] bf16

  const int tid = threadIdx.x;
  const int b = blockIdx.x / HN;
  const int y = blockIdx.x % HN;

  { // head weights -> LDS once
    const uint4* s = (const uint4*)Wh;
    uint4* d = (uint4*)Whs;
    #pragma unroll
    for (int j = 0; j < 4; ++j) d[j * 256 + tid] = s[j * 256 + tid];
  }

  float vmem[48], imem[48];
  #pragma unroll
  for (int j = 0; j < 48; ++j) { vmem[j] = 0.f; imem[j] = 0.f; }

  const int lane = tid & 31, wid = tid >> 5;
  const int l16 = lane & 15, hi = lane >> 4, klo = hi * 8;

  for (int t = 0; t < TSN; ++t) {
    const int n = t * BN + b;
    #pragma unroll
    for (int j = 0; j < 48; ++j) {
      const int f = j * 256 + tid;
      const int c = f / 48, px = f - c * 48;
      const float x = Y[(((size_t)n * CN + c) * HN + y) * WN + px];
      vmem[j] += 0.1f * (imem[j] - vmem[j]);      // v uses previous-step i
      imem[j]  = 0.8f * imem[j] + x;              // i decays then adds input
      Vs[px * CN + c] = f2bf(vmem[j]);
    }
    __syncthreads();

    if (wid < 6) {  // 6 tiles: 2 o-subtiles x 3 px-subtiles
      const int msub = wid & 1, nsub = wid >> 1;
      v8f acc = {0.f,0.f,0.f,0.f,0.f,0.f,0.f,0.f};
      #pragma unroll
      for (int kc = 0; kc < 8; ++kc) {
        const int k0 = kc * 32 + klo;
        const v8bf* pa = (const v8bf*)&Whs[(msub * 16 + l16) * CN + k0];
        const v8bf* pb = (const v8bf*)&Vs [(nsub * 16 + l16) * CN + k0];
        ABu a, bb;
        a.h[0]  = pa[0]; a.h[1]  = pa[2];
        bb.h[0] = pb[0]; bb.h[1] = pb[2];
        acc = __builtin_amdgcn_wmma_f32_16x16x32_bf16(
            false, a.v, false, bb.v, (short)0, acc, false, false);
      }
      const int px = nsub * 16 + l16;
      #pragma unroll
      for (int vv = 0; vv < 8; ++vv) {
        const int o = msub * 16 + hi * 8 + vv;
        if (o < 12) {
          out[((size_t)(t * BN + b) * 12 + o) * (HN * WN) + y * WN + px] =
              acc[vv] + box_b[o];
        } else if (o < 21) {
          out[BOXELT + ((size_t)(t * BN + b) * 9 + (o - 12)) * (HN * WN) + y * WN + px] =
              acc[vv] + cls_b[o - 12];
        }
      }
    }
    __syncthreads();  // protect Vs before next timestep overwrite
  }
}

extern "C" void kernel_launch(void* const* d_in, const int* in_sizes, int n_in,
                              void* d_out, int out_size, void* d_ws, size_t ws_size,
                              hipStream_t stream) {
  const float* X      = (const float*)d_in[0];
  const float* conv_w = (const float*)d_in[1];
  const float* conv_b = (const float*)d_in[2];
  const float* box_w  = (const float*)d_in[3];
  const float* box_b  = (const float*)d_in[4];
  const float* cls_w  = (const float*)d_in[5];
  const float* cls_b  = (const float*)d_in[6];
  float* out = (float*)d_out;

  unsigned short* Bw = (unsigned short*)d_ws;                      // 1.18 MB bf16 weights
  unsigned short* Wh = (unsigned short*)((char*)d_ws + WS_WH);     // 16 KB head pack
  float*          Yb = (float*)((char*)d_ws + WS_Y);               // 151 MB conv output

  prep_conv_w<<<(CN * CN * 9 + 255) / 256, 256, 0, stream>>>(conv_w, Bw);
  prep_head_w<<<32, 256, 0, stream>>>(box_w, cls_w, Wh);
  conv3x3_wmma<<<NIMG * 24, 256, 0, stream>>>(X, Bw, conv_b, Yb);
  li_scan_heads<<<BN * HN, 256, 0, stream>>>(Yb, Wh, box_b, cls_b, out);
}
